// KWS_SNN_87376814670589
// MI455X (gfx1250) — matrix operand order repaired and checked by reading, not compiled, not run on hardware
//
#include <hip/hip_runtime.h>
#include <hip/hip_bf16.h>

// ---------------------------------------------------------------------------
// Types for CDNA5 WMMA (wave32): A/B = 16 bf16 per lane, C/D = 8 f32 per lane
// ---------------------------------------------------------------------------
typedef __attribute__((ext_vector_type(16))) __bf16 v16bf;
typedef __attribute__((ext_vector_type(8)))  float  v8f;
typedef __attribute__((ext_vector_type(4)))  float  vf4;

#define WMMA_BF16(a, b, c) \
    __builtin_amdgcn_wmma_f32_16x16x32_bf16(false, (a), false, (b), (short)0, (c), false, false)

// ---------------------------------------------------------------------------
// WMMA fragment helpers (contiguous-k layouts used by this code, wave32)
// A 16x32 bf16, row-major source with leading dim ld:
//   lane<16: row M=lane,     K = 0..15
//   lane>=16: row M=lane-16, K = 16..31
// ---------------------------------------------------------------------------
static __device__ inline v16bf load_a_rm(const __bf16* base, int ld, int m0, int k0, int lane) {
    const __bf16* row = base + (m0 + (lane & 15)) * ld + k0 + ((lane & 16) ? 16 : 0);
    v16bf a;
#pragma unroll
    for (int i = 0; i < 8; ++i) a[i] = row[i];
#pragma unroll
    for (int i = 0; i < 8; ++i) a[8 + i] = row[8 + i];
    return a;
}

// B 32x16 bf16 taken from a matrix stored as W[N][K] (row-major, ld >= 32):
//   lane = column n; lane<16 holds K 0..15, lane>=16 holds K 16..31
//   -> 32 contiguous bytes per lane (two ds/global b128 loads)
static __device__ inline v16bf load_bT(const __bf16* base, int ld, int n0, int k0, int lane) {
    const __bf16* row = base + (n0 + (lane & 15)) * ld + k0 + ((lane & 16) ? 16 : 0);
    v16bf b;
#pragma unroll
    for (int i = 0; i < 16; ++i) b[i] = row[i];
    return b;
}

// D 16x16 f32: lane<16 rows m0..m0+7, lane>=16 rows m0+8..m0+15; col = lane&15
static __device__ inline void store_d(float* base, int ld, int m0, int n0, int lane, v8f acc) {
    int n  = n0 + (lane & 15);
    int mo = (lane & 16) ? 8 : 0;
#pragma unroll
    for (int j = 0; j < 8; ++j) base[(m0 + mo + j) * ld + n] = acc[j];
}

// ---------------------------------------------------------------------------
// Async global -> LDS bulk copy (CDNA5 GLOBAL_LOAD_ASYNC_TO_LDS, ASYNCcnt).
// lds_off is the absolute LDS byte offset (dynamic-LDS base == 0 in kernels
// with no static __shared__). GVS addressing: saddr base + 32-bit voffset.
// ---------------------------------------------------------------------------
static __device__ inline void async_g2l_b128(unsigned lds_off, const void* gbase, unsigned goff) {
    asm volatile("global_load_async_to_lds_b128 %0, %1, %2 offset:0"
                 :
                 : "v"(lds_off), "v"(goff), "s"(gbase)
                 : "memory");
}
static __device__ inline void wait_async0() {
    asm volatile("s_wait_asynccnt 0x0" ::: "memory");
}

// ---------------------------------------------------------------------------
// Problem constants
// ---------------------------------------------------------------------------
#define T_STEPS 25
#define BATCH   512
#define H_IN    100
#define W_IN    64
#define BHW     (BATCH * H_IN * W_IN)     // 3,276,800
#define FLAT    12800                     // 32*25*16
#define N1      256
#define N2      128
#define N3      35

// ---------------------------------------------------------------------------
// K1: time-mean over T (pure bandwidth). x is streamed once (328 MB) -> use
// non-temporal loads so it does not evict weights/activations from L2.
// ---------------------------------------------------------------------------
__global__ void mean_kernel(const float* __restrict__ x, float* __restrict__ xm) {
    int i = blockIdx.x * blockDim.x + threadIdx.x;       // over BHW/4
    const int n4 = BHW / 4;
    if (i >= n4) return;
    const vf4* xin = (const vf4*)x;
    vf4 acc = {0.f, 0.f, 0.f, 0.f};
#pragma unroll 5
    for (int t = 0; t < T_STEPS; ++t) {
        vf4 v = __builtin_nontemporal_load(&xin[t * n4 + i]);
        acc += v;
    }
    ((vf4*)xm)[i] = acc * (1.0f / (float)T_STEPS);
}

// ---------------------------------------------------------------------------
// K2: one-shot weight conversions to bf16 (fc1, fc2, fc3-padded, conv2-padded)
// ---------------------------------------------------------------------------
#define CW_N1 (256L * 12800L)   // fc1_w
#define CW_N2 (128L * 256L)     // fc2_w
#define CW_N3 (48L * 128L)      // fc3_w padded rows 35..47 -> 0
#define CW_N4 (32L * 160L)      // conv2_w [oc][k], k=ic*9+kh*3+kw, padded 144->160
#define CW_TOTAL (CW_N1 + CW_N2 + CW_N3 + CW_N4)

__global__ void convw_kernel(const float* __restrict__ fc1w, const float* __restrict__ fc2w,
                             const float* __restrict__ fc3w, const float* __restrict__ c2w,
                             __bf16* o1, __bf16* o2, __bf16* o3, __bf16* o4) {
    long i = (long)blockIdx.x * blockDim.x + threadIdx.x;
    if (i < CW_N1) { o1[i] = (__bf16)fc1w[i]; return; }
    i -= CW_N1;
    if (i < CW_N2) { o2[i] = (__bf16)fc2w[i]; return; }
    i -= CW_N2;
    if (i < CW_N3) {
        int n = (int)(i / 128), k = (int)(i % 128);
        o3[i] = (__bf16)((n < N3) ? fc3w[n * 128 + k] : 0.f);
        return;
    }
    i -= CW_N3;
    if (i < CW_N4) {
        int oc = (int)(i / 160), k = (int)(i % 160);
        o4[i] = (__bf16)((k < 144) ? c2w[oc * 144 + k] : 0.f);
    }
}

// ---------------------------------------------------------------------------
// K3: conv1(1->16,3x3,pad1) + BN1 + ReLU + maxpool2 -> p1 (B,16,50,32) bf16
// ---------------------------------------------------------------------------
__global__ void conv1_kernel(const float* __restrict__ xm, const float* __restrict__ wgt,
                             const float* __restrict__ cb,
                             const float* __restrict__ g, const float* __restrict__ bb,
                             const float* __restrict__ bm, const float* __restrict__ bv,
                             __bf16* __restrict__ p1) {
    int id = blockIdx.x * blockDim.x + threadIdx.x;      // B*16*50*32 = 13107200
    if (id >= BATCH * 16 * 50 * 32) return;
    int pw = id & 31;
    int t  = id >> 5;
    int ph = t % 50; t /= 50;
    int c  = t & 15;
    int b  = t >> 4;

    float inv  = g[c] * rsqrtf(bv[c] + 1e-5f);
    float bias = cb[c] * inv + bb[c] - bm[c] * inv;

    float w[9];
#pragma unroll
    for (int k = 0; k < 9; ++k) w[k] = wgt[c * 9 + k];

    const float* img = xm + b * (H_IN * W_IN);
    float best = 0.f;  // ReLU output is >= 0
#pragma unroll
    for (int dy = 0; dy < 2; ++dy) {
#pragma unroll
        for (int dx = 0; dx < 2; ++dx) {
            int oh = 2 * ph + dy, ow = 2 * pw + dx;
            float acc = 0.f;
#pragma unroll
            for (int kh = 0; kh < 3; ++kh) {
#pragma unroll
                for (int kw = 0; kw < 3; ++kw) {
                    int ih = oh - 1 + kh, iw = ow - 1 + kw;
                    if (ih >= 0 && ih < H_IN && iw >= 0 && iw < W_IN)
                        acc += img[ih * W_IN + iw] * w[kh * 3 + kw];
                }
            }
            float v = fmaxf(0.f, acc * inv + bias);
            best = fmaxf(best, v);
        }
    }
    p1[id] = (__bf16)best;
}

// ---------------------------------------------------------------------------
// K4: conv2(16->32,3x3,pad1) + BN2 + ReLU + maxpool2 via WMMA implicit GEMM
// One block per (batch, pooled-row). im2col tile stored TRANSPOSED in LDS as
// [spatial=64][K=160] with pitch 168 (conflict-free, contiguous b128 B-frags).
// Weights [32][K] pitch 168. 8 waves x (2x4) 16x16 tiles x 5 K-chunks.
// ---------------------------------------------------------------------------
#define C2_LDW 168   // bf16 pitch for sW / sColT (84 dwords -> no bank conflicts)
#define C2_LDO 68    // f32 pitch for sO

__global__ void conv2_wmma_kernel(const __bf16* __restrict__ p1, const __bf16* __restrict__ w2,
                                  const float* __restrict__ c2b,
                                  const float* __restrict__ g2, const float* __restrict__ bb2,
                                  const float* __restrict__ bm2, const float* __restrict__ bv2,
                                  __bf16* __restrict__ feat) {
    __shared__ __bf16 sW[32 * C2_LDW];
    __shared__ __bf16 sColT[64 * C2_LDW];
    __shared__ float  sO[32 * C2_LDO];

    int tid  = threadIdx.x;
    int lane = tid & 31;
    int wv   = tid >> 5;
    int b    = blockIdx.x / 25;
    int pr   = blockIdx.x % 25;          // pooled row -> conv rows 2*pr, 2*pr+1

    for (int i = tid; i < 32 * 160; i += 256) sW[(i / 160) * C2_LDW + (i % 160)] = w2[i];

    for (int e = tid; e < 160 * 64; e += 256) {
        int k = e >> 6, s = e & 63;
        __bf16 v = (__bf16)0.f;
        if (k < 144) {
            int ic = k / 9, r = k % 9, kh = r / 3, kw = r - kh * 3;
            int oh = 2 * pr + (s >> 5), ow = s & 31;
            int ih = oh - 1 + kh, iw = ow - 1 + kw;
            if (ih >= 0 && ih < 50 && iw >= 0 && iw < 32)
                v = p1[((b * 16 + ic) * 50 + ih) * 32 + iw];
        }
        sColT[s * C2_LDW + k] = v;       // transposed store: [s][k]
    }
    __syncthreads();

    int mt = wv >> 2, nt = wv & 3;       // 2 x 4 tile grid over [32 oc][64 spatial]
    v8f acc = {};
#pragma unroll
    for (int kc = 0; kc < 5; ++kc) {
        v16bf a  = load_a_rm(sW,    C2_LDW, mt * 16, kc * 32, lane);
        v16bf bq = load_bT (sColT,  C2_LDW, nt * 16, kc * 32, lane);
        acc = WMMA_BF16(a, bq, acc);
    }
    store_d(sO, C2_LDO, mt * 16, nt * 16, lane, acc);
    __syncthreads();

    for (int e = tid; e < 512; e += 256) {
        int oc = e >> 4, pw = e & 15;
        float inv  = g2[oc] * rsqrtf(bv2[oc] + 1e-5f);
        float bias = c2b[oc] * inv + bb2[oc] - bm2[oc] * inv;
        const float* row = sO + oc * C2_LDO;
        float v00 = fmaxf(0.f, row[2 * pw]          * inv + bias);
        float v01 = fmaxf(0.f, row[2 * pw + 1]      * inv + bias);
        float v10 = fmaxf(0.f, row[32 + 2 * pw]     * inv + bias);
        float v11 = fmaxf(0.f, row[32 + 2 * pw + 1] * inv + bias);
        float pv  = fmaxf(fmaxf(v00, v01), fmaxf(v10, v11));
        feat[b * FLAT + oc * 400 + pr * 16 + pw] = (__bf16)pv;
    }
}

// ---------------------------------------------------------------------------
// K5: fc1 WMMA GEMM: cur1[512,256] = feat[512,12800] @ fc1_w[256,12800]^T + b
// 512 wave-tiles (32 M-tiles x 16 N-tiles), 400 K-chunks; operands live in L2.
// ---------------------------------------------------------------------------
__global__ void fc1_wmma_kernel(const __bf16* __restrict__ feat, const __bf16* __restrict__ w1,
                                const float* __restrict__ b1, float* __restrict__ cur1) {
    int lane = threadIdx.x & 31;
    int gw   = blockIdx.x * 8 + (threadIdx.x >> 5);
    int m0   = (gw >> 4) * 16;
    int n0   = (gw & 15) * 16;
    v8f acc  = {};
#pragma unroll 4
    for (int kc = 0; kc < FLAT / 32; ++kc) {
        v16bf a = load_a_rm(feat, FLAT, m0, kc * 32, lane);
        v16bf b = load_bT (w1,   FLAT, n0, kc * 32, lane);
        acc = WMMA_BF16(a, b, acc);
    }
    int n  = n0 + (lane & 15);
    int mo = (lane & 16) ? 8 : 0;
    float bias = b1[n];
#pragma unroll
    for (int j = 0; j < 8; ++j) cur1[(m0 + mo + j) * N1 + n] = acc[j] + bias;
}

// ---------------------------------------------------------------------------
// K6: persistent LIF temporal loop. 16 blocks x 32 batch rows x 256 threads.
// fc2/fc3 weights staged once via ASYNC global->LDS DMA; membranes live in
// registers; per step: LIF1 -> WMMA (32x128x256) -> LIF2 -> WMMA (32x48x128)
// -> LIF3 -> spike output. All LDS pitches padded for conflict-free frags.
//
// Dynamic-LDS layout (byte offsets; base of dynamic LDS == LDS addr 0):
//   sW2  @      0 : 128 x 264 bf16 (data 256/row, pitch 528B)  67,584
//   sW3  @  67584 :  48 x 136 bf16 (data 128/row, pitch 272B)  13,056
//   sS1  @  80640 :  32 x 264 bf16                             16,896
//   sS2  @  97536 :  32 x 136 bf16                              8,704
//   sZ2  @ 106240 :  32 x 132 f32                              16,896
//   sZ3  @ 123136 :  32 x  52 f32                               6,656
//   total 129,792
// ---------------------------------------------------------------------------
#define LD_W2 264
#define LD_W3 136
#define LD_S1 264
#define LD_S2 136
#define LD_Z2 132
#define LD_Z3 52
#define OFF_W3 67584u
#define OFF_S1 80640u
#define OFF_S2 97536u
#define OFF_Z2 106240u
#define OFF_Z3 123136u
#define LIF_SMEM_BYTES 129792

__global__ void lif_persistent_kernel(const float* __restrict__ cur1,
                                      const __bf16* __restrict__ w2, const __bf16* __restrict__ w3,
                                      const float* __restrict__ b2, const float* __restrict__ b3,
                                      const float* __restrict__ pb1, const float* __restrict__ pb2,
                                      const float* __restrict__ pb3,
                                      float* __restrict__ out) {
    extern __shared__ char smem[];
    __bf16* sW2 = (__bf16*)(smem);
    __bf16* sW3 = (__bf16*)(smem + OFF_W3);
    __bf16* sS1 = (__bf16*)(smem + OFF_S1);
    __bf16* sS2 = (__bf16*)(smem + OFF_S2);
    float*  sZ2 = (float*) (smem + OFF_Z2);
    float*  sZ3 = (float*) (smem + OFF_Z3);

    int tid  = threadIdx.x;
    int lane = tid & 31;
    int wv   = tid >> 5;
    int b0   = blockIdx.x * 32;

    // ---- async DMA: fc2_w (128 rows x 512B) and fc3_w (48 rows x 256B) ----
    for (int c = tid; c < 4096; c += 256) {               // 128*32 b128 chunks
        unsigned row = (unsigned)c >> 5, col = ((unsigned)c & 31) * 16;
        async_g2l_b128(row * 528 + col, w2, row * 512 + col);
    }
    for (int c = tid; c < 768; c += 256) {                // 48*16 b128 chunks
        unsigned row = (unsigned)c >> 4, col = ((unsigned)c & 15) * 16;
        async_g2l_b128(OFF_W3 + row * 272 + col, w3, row * 256 + col);
    }

    float beta1 = fminf(fmaxf(pb1[0], 0.f), 1.f);
    float beta2 = fminf(fmaxf(pb2[0], 0.f), 1.f);
    float beta3 = fminf(fmaxf(pb3[0], 0.f), 1.f);

    // layer-1 state: thread owns batch row bb, 32 neurons starting at c0
    int bb = tid >> 3, c0 = (tid & 7) * 32;
    float m1r[32], c1r[32];
#pragma unroll
    for (int i = 0; i < 32; ++i) { c1r[i] = cur1[(b0 + bb) * N1 + c0 + i]; m1r[i] = 0.f; }

    // layer-2 state: same batch row, 16 neurons starting at n2
    int n2 = (tid & 7) * 16;
    float m2r[16], b2r[16];
#pragma unroll
    for (int i = 0; i < 16; ++i) { b2r[i] = b2[n2 + i]; m2r[i] = 0.f; }

    // layer-3 state: batch row b3i, up to 5 neurons
    int b3i = tid & 31, j3 = tid >> 5;
    float m3r[5], b3r[5];
#pragma unroll
    for (int i = 0; i < 5; ++i) {
        int n = j3 * 5 + i;
        b3r[i] = (n < N3) ? b3[n] : 0.f;
        m3r[i] = 0.f;
    }

    wait_async0();          // weights resident in LDS
    __syncthreads();

    for (int t = 0; t < T_STEPS; ++t) {
        // ---- LIF layer 1 (reset from old mem, spike from new mem) ----
#pragma unroll
        for (int i = 0; i < 32; ++i) {
            float rst = (m1r[i] > 1.f) ? 1.f : 0.f;
            m1r[i] = beta1 * m1r[i] + c1r[i] - rst;
            sS1[bb * LD_S1 + c0 + i] = (__bf16)((m1r[i] > 1.f) ? 1.f : 0.f);
        }
        __syncthreads();

        // ---- z2 = spk1 @ fc2_w^T : M=32, N=128, K=256; wave wv owns N-tile wv ----
#pragma unroll
        for (int mt = 0; mt < 2; ++mt) {
            v8f acc = {};
#pragma unroll
            for (int kc = 0; kc < 8; ++kc) {
                v16bf a  = load_a_rm(sS1, LD_S1, mt * 16, kc * 32, lane);
                v16bf bq = load_bT (sW2, LD_W2, wv * 16, kc * 32, lane);
                acc = WMMA_BF16(a, bq, acc);
            }
            store_d(sZ2, LD_Z2, mt * 16, wv * 16, lane, acc);
        }
        __syncthreads();

        // ---- LIF layer 2 ----
#pragma unroll
        for (int i = 0; i < 16; ++i) {
            float z   = sZ2[bb * LD_Z2 + n2 + i] + b2r[i];
            float rst = (m2r[i] > 1.f) ? 1.f : 0.f;
            m2r[i] = beta2 * m2r[i] + z - rst;
            sS2[bb * LD_S2 + n2 + i] = (__bf16)((m2r[i] > 1.f) ? 1.f : 0.f);
        }
        __syncthreads();

        // ---- z3 = spk2 @ fc3_w^T : M=32, N=48(pad), K=128; waves 0..5 ----
        if (wv < 6) {
            int mt = wv / 3, nt = wv % 3;
            v8f acc = {};
#pragma unroll
            for (int kc = 0; kc < 4; ++kc) {
                v16bf a  = load_a_rm(sS2, LD_S2, mt * 16, kc * 32, lane);
                v16bf bq = load_bT (sW3, LD_W3, nt * 16, kc * 32, lane);
                acc = WMMA_BF16(a, bq, acc);
            }
            store_d(sZ3, LD_Z3, mt * 16, nt * 16, lane, acc);
        }
        __syncthreads();

        // ---- LIF layer 3 + spike output ----
#pragma unroll
        for (int i = 0; i < 5; ++i) {
            int n = j3 * 5 + i;
            if (n < N3) {
                float z   = sZ3[b3i * LD_Z3 + n] + b3r[i];
                float rst = (m3r[i] > 1.f) ? 1.f : 0.f;
                m3r[i] = beta3 * m3r[i] + z - rst;
                out[((long)t * BATCH + b0 + b3i) * N3 + n] = (m3r[i] > 1.f) ? 1.f : 0.f;
            }
        }
        __syncthreads();
    }
}

// ---------------------------------------------------------------------------
// Launch. Workspace layout (bytes, 256-aligned):
//   0         xmean  f32   13,107,200
//   13107200  p1     bf16  26,214,400
//   39321600  feat   bf16  13,107,200
//   52428800  cur1   f32      524,288
//   52953088  fc1w   bf16   6,553,600
//   59506688  fc2w   bf16      65,536
//   59572224  fc3w   bf16      12,288
//   59584512  w2pad  bf16      10,240   (end 59,594,752)
// ---------------------------------------------------------------------------
extern "C" void kernel_launch(void* const* d_in, const int* in_sizes, int n_in,
                              void* d_out, int out_size, void* d_ws, size_t ws_size,
                              hipStream_t stream) {
    const float* x     = (const float*)d_in[0];
    const float* c1w   = (const float*)d_in[1];
    const float* c1b   = (const float*)d_in[2];
    const float* bn1g  = (const float*)d_in[3];
    const float* bn1b  = (const float*)d_in[4];
    const float* bn1m  = (const float*)d_in[5];
    const float* bn1v  = (const float*)d_in[6];
    const float* c2w   = (const float*)d_in[7];
    const float* c2b   = (const float*)d_in[8];
    const float* bn2g  = (const float*)d_in[9];
    const float* bn2b  = (const float*)d_in[10];
    const float* bn2m  = (const float*)d_in[11];
    const float* bn2v  = (const float*)d_in[12];
    const float* fc1w  = (const float*)d_in[13];
    const float* fc1b  = (const float*)d_in[14];
    const float* fc2w  = (const float*)d_in[15];
    const float* fc2b  = (const float*)d_in[16];
    const float* fc3w  = (const float*)d_in[17];
    const float* fc3b  = (const float*)d_in[18];
    const float* pb1   = (const float*)d_in[19];
    const float* pb2   = (const float*)d_in[20];
    const float* pb3   = (const float*)d_in[21];

    char* ws = (char*)d_ws;
    float*  xmean = (float*) (ws + 0);
    __bf16* p1    = (__bf16*)(ws + 13107200);
    __bf16* feat  = (__bf16*)(ws + 39321600);
    float*  cur1  = (float*) (ws + 52428800);
    __bf16* w1q   = (__bf16*)(ws + 52953088);
    __bf16* w2q   = (__bf16*)(ws + 59506688);
    __bf16* w3q   = (__bf16*)(ws + 59572224);
    __bf16* c2q   = (__bf16*)(ws + 59584512);

    mean_kernel<<<(BHW / 4 + 255) / 256, 256, 0, stream>>>(x, xmean);
    convw_kernel<<<(int)((CW_TOTAL + 255) / 256), 256, 0, stream>>>(
        fc1w, fc2w, fc3w, c2w, w1q, w2q, w3q, c2q);
    conv1_kernel<<<(BATCH * 16 * 50 * 32 + 255) / 256, 256, 0, stream>>>(
        xmean, c1w, c1b, bn1g, bn1b, bn1m, bn1v, p1);
    conv2_wmma_kernel<<<BATCH * 25, 256, 0, stream>>>(
        p1, c2q, c2b, bn2g, bn2b, bn2m, bn2v, feat);
    fc1_wmma_kernel<<<64, 256, 0, stream>>>(feat, w1q, fc1b, cur1);
    lif_persistent_kernel<<<16, 256, LIF_SMEM_BYTES, stream>>>(
        cur1, w2q, w3q, fc2b, fc3b, pb1, pb2, pb3, (float*)d_out);
}
